// DNC_65189013619263
// MI455X (gfx1250) — matrix operand order repaired and checked by reading, not compiled, not run on hardware
//
#include <hip/hip_runtime.h>
#include <hip/hip_bf16.h>

// ---------------- model constants ----------------
#define VOCAB   32000
#define DMODEL  1024
#define NLAYER  6
#define NHEAD   16
#define DHEAD   64
#define NSLOT   4096
#define DFF     4096
#define TSEQ    2048
#define NBATCH  2
#define NTOK    (NBATCH*TSEQ)   // 4096 token rows

typedef __bf16 bf16_t;
typedef __attribute__((ext_vector_type(16))) bf16_t bf16x16;
typedef __attribute__((ext_vector_type(8)))  bf16_t bf16x8;
typedef __attribute__((ext_vector_type(8)))  float  f32x8;

// round-to-nearest-even f32 -> bf16
__device__ __forceinline__ bf16_t f2bf(float f) {
    union { float f; unsigned u; } in; in.f = f;
    unsigned u = in.u;
    u += 0x7FFFu + ((u >> 16) & 1u);
    unsigned short h = (unsigned short)(u >> 16);
    union { unsigned short s; bf16_t b; } out; out.s = h;
    return out.b;
}

// ---- WMMA fragment loaders (ISA 16-bit 16x32 A layout / 32x16 B layout) ----
// A: lane m = lane%16; e<8 -> K = half*8+e ; e>=8 -> K = 16+half*8+(e-8)
__device__ __forceinline__ bf16x16 load_frag_a(const bf16_t* __restrict__ arow,
                                               int k0, int half) {
    const bf16_t* ap = arow + k0 + half * 8;
    bf16x8 lo = *(const bf16x8*)(ap);
    bf16x8 hi = *(const bf16x8*)(ap + 16);
    bf16x16 a;
    #pragma unroll
    for (int e = 0; e < 8; ++e) { a[e] = lo[e]; a[8 + e] = hi[e]; }
    return a;
}
// B (from B^T row for this lane's column): e<8 -> K = half*8+e ; e>=8 -> K = 16+half*8+(e-8)
__device__ __forceinline__ bf16x16 load_frag_b(const bf16_t* __restrict__ brow,
                                               int k0, int half) {
    const bf16_t* bp = brow + k0 + half * 8;
    bf16x8 lo = *(const bf16x8*)(bp);
    bf16x8 hi = *(const bf16x8*)(bp + 16);
    bf16x16 b;
    #pragma unroll
    for (int e = 0; e < 8; ++e) { b[e] = lo[e]; b[8 + e] = hi[e]; }
    return b;
}

// ---------------- f32 -> bf16 bulk convert (same layout) ----------------
__global__ void k_cvt(const float* __restrict__ s, bf16_t* __restrict__ d, size_t n) {
    size_t i = (size_t)blockIdx.x * blockDim.x + threadIdx.x;
    size_t stride = (size_t)gridDim.x * blockDim.x;
    for (; i < n; i += stride) d[i] = f2bf(s[i]);
}

// ---------------- f32 [R,C] -> bf16 [C,R] tiled transpose-convert ----------------
// block (32,8), grid (C/32, R/32). Requires R%32==0, C%32==0.
__global__ void k_cvt_t(const float* __restrict__ src, bf16_t* __restrict__ dst,
                        int R, int C) {
    __shared__ float tile[32][33];
    const int c0 = blockIdx.x * 32;
    const int r0 = blockIdx.y * 32;
    const int tx = threadIdx.x, ty = threadIdx.y;
    #pragma unroll
    for (int j = 0; j < 32; j += 8)
        tile[ty + j][tx] = src[(size_t)(r0 + ty + j) * C + (c0 + tx)];
    __syncthreads();
    #pragma unroll
    for (int j = 0; j < 32; j += 8)
        dst[(size_t)(c0 + ty + j) * R + (r0 + tx)] = f2bf(tile[tx][ty + j]);
}

// ---------------- embedding gather + RoPE (writes f32 + bf16) ----------------
__global__ void k_embed_rope(const int* __restrict__ ids, const float* __restrict__ emb,
                             float* __restrict__ xf, bf16_t* __restrict__ xb) {
    int tok = blockIdx.x;               // 0..NTOK-1
    int t   = tok & (TSEQ - 1);
    int id  = ids[tok];
    const float* e  = emb + (size_t)id * DMODEL;
    float*  xo  = xf + (size_t)tok * DMODEL;
    bf16_t* xbo = xb + (size_t)tok * DMODEL;
    for (int p = threadIdx.x; p < DMODEL / 2; p += blockDim.x) {
        float inv = __expf(-(2.0f * (float)p / (float)DMODEL) * 9.210340371976184f);
        float fr = (float)t * inv;
        float c = cosf(fr), s = sinf(fr);
        float x1 = e[2*p], x2 = e[2*p+1];
        float o0 = x1 * c - x2 * s;
        float o1 = x1 * s + x2 * c;
        xo[2*p] = o0;  xo[2*p+1] = o1;
        xbo[2*p] = f2bf(o0);  xbo[2*p+1] = f2bf(o1);
    }
}

// ---------------- bf16 WMMA GEMM, B pre-transposed, register double-buffered ----
// C[M,N] = alpha * A[M,K] @ B + bias, where BT holds B^T: [N,K] row-major.
// Block = 128 threads (4 waves). Each wave computes a 16(M) x 64(N) slab as
// 4 accumulators; one A fragment reused by 4 WMMAs per k-step. Next k-step's
// fragments are loaded into a second register set before the current WMMAs
// issue, so loads stay in flight across the matrix ops.
// Block tile = 16 x 256. Requires M%16==0, N%256==0, K%32==0. No LDS/barriers.
__global__ void k_gemm_bf16(const bf16_t* __restrict__ A, const bf16_t* __restrict__ BT,
                            const float* __restrict__ bias, float* __restrict__ C,
                            int M, int N, int K, float alpha) {
    const int lane = threadIdx.x & 31;
    const int wave = threadIdx.x >> 5;          // 0..3
    const int half = lane >> 4;                 // 0/1
    const int lm   = lane & 15;
    const int m0   = blockIdx.y * 16;
    const int n0   = (blockIdx.x * 4 + wave) * 64;   // wave's 64-wide slab

    f32x8 acc[4] = {};
    const bf16_t* arow = A  + (size_t)(m0 + lm) * K;
    const bf16_t* brow[4];
    #pragma unroll
    for (int t = 0; t < 4; ++t)
        brow[t] = BT + (size_t)(n0 + 16 * t + lm) * K;

    // stage 0
    bf16x16 a_cur = load_frag_a(arow, 0, half);
    bf16x16 b_cur[4];
    #pragma unroll
    for (int t = 0; t < 4; ++t) b_cur[t] = load_frag_b(brow[t], 0, half);

    for (int k0 = 0; k0 < K; k0 += 32) {
        const int kn = (k0 + 32 < K) ? (k0 + 32) : 0;   // last iter: dummy in-bounds
        bf16x16 a_nxt = load_frag_a(arow, kn, half);
        bf16x16 b_nxt[4];
        #pragma unroll
        for (int t = 0; t < 4; ++t) b_nxt[t] = load_frag_b(brow[t], kn, half);

        #pragma unroll
        for (int t = 0; t < 4; ++t)
            acc[t] = __builtin_amdgcn_wmma_f32_16x16x32_bf16(
                         false, a_cur, false, b_cur[t], (short)0, acc[t], false, false);

        a_cur = a_nxt;
        #pragma unroll
        for (int t = 0; t < 4; ++t) b_cur[t] = b_nxt[t];
    }

    #pragma unroll
    for (int t = 0; t < 4; ++t) {
        const int ncol = n0 + 16 * t + lm;
        float bv = bias ? bias[ncol] : 0.0f;
        #pragma unroll
        for (int r = 0; r < 8; ++r) {           // C/D layout: VGPR r -> row r + 8*half
            int m = m0 + r + 8 * half;
            C[(size_t)m * N + ncol] = alpha * acc[t][r] + bv;
        }
    }
}

// ---------------- flash attention (one thread per (b,h,q) row) ----------------
__global__ void k_attn(const float* __restrict__ Q, const float* __restrict__ Km,
                       const float* __restrict__ Vm, bf16_t* __restrict__ O) {
    int gid = blockIdx.x * blockDim.x + threadIdx.x;     // NTOK*NHEAD threads
    int b   = gid / (NHEAD * TSEQ);
    int rem = gid % (NHEAD * TSEQ);
    int h   = rem / TSEQ;
    int qi  = rem % TSEQ;
    const size_t base = (size_t)b * TSEQ * DMODEL + (size_t)h * DHEAD;

    const float4* qp = (const float4*)(Q + base + (size_t)qi * DMODEL);
    float4 qv[16];
    #pragma unroll
    for (int j = 0; j < 16; ++j) qv[j] = qp[j];

    float mx = -1e30f, l = 0.f;
    float4 acc[16];
    #pragma unroll
    for (int j = 0; j < 16; ++j) acc[j] = make_float4(0.f, 0.f, 0.f, 0.f);

    for (int ki = 0; ki < TSEQ; ++ki) {
        const float4* kp = (const float4*)(Km + base + (size_t)ki * DMODEL);
        float s = 0.f;
        #pragma unroll
        for (int j = 0; j < 16; ++j) {
            float4 kk = kp[j];
            s += qv[j].x * kk.x + qv[j].y * kk.y + qv[j].z * kk.z + qv[j].w * kk.w;
        }
        s *= 0.125f;                                      // DH^-0.5
        float nm   = fmaxf(mx, s);
        float corr = __expf(mx - nm);
        float p    = __expf(s - nm);
        l = l * corr + p;
        const float4* vp = (const float4*)(Vm + base + (size_t)ki * DMODEL);
        #pragma unroll
        for (int j = 0; j < 16; ++j) {
            float4 vv = vp[j];
            acc[j].x = acc[j].x * corr + p * vv.x;
            acc[j].y = acc[j].y * corr + p * vv.y;
            acc[j].z = acc[j].z * corr + p * vv.z;
            acc[j].w = acc[j].w * corr + p * vv.w;
        }
        mx = nm;
    }
    float invl = 1.f / l;
    bf16_t* op = O + base + (size_t)qi * DMODEL;
    #pragma unroll
    for (int j = 0; j < 16; ++j) {
        op[4*j+0] = f2bf(acc[j].x * invl);
        op[4*j+1] = f2bf(acc[j].y * invl);
        op[4*j+2] = f2bf(acc[j].z * invl);
        op[4*j+3] = f2bf(acc[j].w * invl);
    }
}

// ---------------- y = rmsnorm(x + t, w); writes f32 + bf16 ----------------
__global__ void k_resid_rmsnorm(const float* __restrict__ X, const float* __restrict__ Tt,
                                const float* __restrict__ W, float* __restrict__ Y,
                                bf16_t* __restrict__ Yb) {
    __shared__ float red[256];
    int row = blockIdx.x;
    const float* xr = X  + (size_t)row * DMODEL;
    const float* tr = Tt + (size_t)row * DMODEL;
    float s[4]; float ss = 0.f;
    #pragma unroll
    for (int j = 0; j < 4; ++j) {
        int c = threadIdx.x + 256 * j;
        s[j] = xr[c] + tr[c];
        ss += s[j] * s[j];
    }
    red[threadIdx.x] = ss;
    __syncthreads();
    for (int st = 128; st > 0; st >>= 1) {
        if (threadIdx.x < st) red[threadIdx.x] += red[threadIdx.x + st];
        __syncthreads();
    }
    float inv = rsqrtf(red[0] / (float)DMODEL + 1e-8f);
    #pragma unroll
    for (int j = 0; j < 4; ++j) {
        int c = threadIdx.x + 256 * j;
        float y = s[j] * inv * W[c];
        Y[(size_t)row * DMODEL + c]  = y;
        Yb[(size_t)row * DMODEL + c] = f2bf(y);
    }
}

// ---------------- row softmax -> bf16 ----------------
__global__ void k_softmax_bf16(const float* __restrict__ S, bf16_t* __restrict__ P, int cols) {
    __shared__ float red[256];
    int row = blockIdx.x;
    const float* sr = S + (size_t)row * cols;
    float mx = -1e30f;
    for (int c = threadIdx.x; c < cols; c += 256) mx = fmaxf(mx, sr[c]);
    red[threadIdx.x] = mx;
    __syncthreads();
    for (int st = 128; st > 0; st >>= 1) {
        if (threadIdx.x < st) red[threadIdx.x] = fmaxf(red[threadIdx.x], red[threadIdx.x + st]);
        __syncthreads();
    }
    mx = red[0];
    __syncthreads();
    float sum = 0.f;
    for (int c = threadIdx.x; c < cols; c += 256) sum += __expf(sr[c] - mx);
    red[threadIdx.x] = sum;
    __syncthreads();
    for (int st = 128; st > 0; st >>= 1) {
        if (threadIdx.x < st) red[threadIdx.x] += red[threadIdx.x + st];
        __syncthreads();
    }
    float inv = 1.f / red[0];
    for (int c = threadIdx.x; c < cols; c += 256)
        P[(size_t)row * cols + c] = f2bf(__expf(sr[c] - mx) * inv);
}

// ---------------- e = silu(g) * u -> bf16 ----------------
__global__ void k_silu_mul(const float* __restrict__ g, const float* __restrict__ u,
                           bf16_t* __restrict__ e, size_t n) {
    size_t i = (size_t)blockIdx.x * blockDim.x + threadIdx.x;
    size_t stride = (size_t)gridDim.x * blockDim.x;
    for (; i < n; i += stride) {
        float gv = g[i];
        e[i] = f2bf((gv / (1.0f + __expf(-gv))) * u[i]);
    }
}

// ==================================================================
extern "C" void kernel_launch(void* const* d_in, const int* in_sizes, int n_in,
                              void* d_out, int out_size, void* d_ws, size_t ws_size,
                              hipStream_t stream) {
    (void)in_sizes; (void)n_in; (void)out_size; (void)ws_size;

    const int*   ids  = (const int*)  d_in[0];
    const float* emb  = (const float*)d_in[1];
    const float* Wq   = (const float*)d_in[2];
    const float* Wk   = (const float*)d_in[3];
    const float* Wv   = (const float*)d_in[4];
    const float* Wo   = (const float*)d_in[5];
    const float* bo   = (const float*)d_in[6];
    const float* n1   = (const float*)d_in[7];
    const float* Wg   = (const float*)d_in[8];
    const float* Wu   = (const float*)d_in[9];
    const float* Wd   = (const float*)d_in[10];
    const float* n2   = (const float*)d_in[11];
    const float* mem  = (const float*)d_in[12];
    const float* Wsp  = (const float*)d_in[13];
    const float* bsp  = (const float*)d_in[14];
    const float* Wrp  = (const float*)d_in[15];
    const float* brp  = (const float*)d_in[16];
    const float* nout = (const float*)d_in[17];

    // deterministic bump allocator over d_ws
    char* wsbase = (char*)d_ws;
    size_t off = 0;
    auto take = [&](size_t bytes) -> void* {
        void* p = wsbase + off;
        off += (bytes + 255) & ~(size_t)255;
        return p;
    };
    const size_t DD = (size_t)DMODEL * DMODEL;

    // bf16 weight shadows -- ALL stored as B^T layout [N,K] for the GEMM,
    // except emb_b / mem_b which are natural [V,D] / [SLOTS,D] (already B^T
    // for the h@emb^T and q@mem^T products).
    bf16_t* emb_b  = (bf16_t*)take((size_t)VOCAB * DMODEL * 2);       // [V, D]
    bf16_t* wqT    = (bf16_t*)take((size_t)NLAYER * DD * 2);          // [D, D] per layer
    bf16_t* wkT    = (bf16_t*)take((size_t)NLAYER * DD * 2);
    bf16_t* wvT    = (bf16_t*)take((size_t)NLAYER * DD * 2);
    bf16_t* woT    = (bf16_t*)take((size_t)NLAYER * DD * 2);
    bf16_t* wgT    = (bf16_t*)take((size_t)NLAYER * DMODEL * DFF * 2);// [FF, D] per layer
    bf16_t* wuT    = (bf16_t*)take((size_t)NLAYER * DMODEL * DFF * 2);
    bf16_t* wdT    = (bf16_t*)take((size_t)NLAYER * DFF * DMODEL * 2);// [D, FF] per layer
    bf16_t* mem_b  = (bf16_t*)take((size_t)NSLOT * DMODEL * 2);       // [SLOTS, D]
    bf16_t* memT   = (bf16_t*)take((size_t)NSLOT * DMODEL * 2);       // [D, SLOTS]
    bf16_t* wspT   = (bf16_t*)take(DD * 2);
    bf16_t* wrpT   = (bf16_t*)take(DD * 2);

    // activations
    float*  xf  = (float*) take((size_t)NTOK * DMODEL * 4);
    bf16_t* xb  = (bf16_t*)take((size_t)NTOK * DMODEL * 2);
    float*  qf  = (float*) take((size_t)NTOK * DMODEL * 4);
    float*  kf  = (float*) take((size_t)NTOK * DMODEL * 4);
    float*  vf  = (float*) take((size_t)NTOK * DMODEL * 4);
    bf16_t* obf = (bf16_t*)take((size_t)NTOK * DMODEL * 2);   // reused for qs_b / r_b
    float*  tf  = (float*) take((size_t)NTOK * DMODEL * 4);   // GEMM temp
    float*  gf  = (float*) take((size_t)NTOK * DFF * 4);
    float*  uf  = (float*) take((size_t)NTOK * DFF * 4);
    bf16_t* ebf = (bf16_t*)take((size_t)NTOK * DFF * 2);
    float*  scf = (float*) take((size_t)NTOK * NSLOT * 4);
    bf16_t* scb = (bf16_t*)take((size_t)NTOK * NSLOT * 2);

    auto cvt = [&](const float* s, bf16_t* d, size_t n) {
        k_cvt<<<4096, 256, 0, stream>>>(s, d, n);
    };
    // src f32 [R,C] -> dst bf16 [C,R]
    auto cvtT = [&](const float* s, bf16_t* d, int R, int C) {
        dim3 grid(C / 32, R / 32), blk(32, 8);
        k_cvt_t<<<grid, blk, 0, stream>>>(s, d, R, C);
    };
    // C[M,N] = alpha*A@B + bias, BT = B^T [N,K]
    auto gemm = [&](const bf16_t* A, const bf16_t* BT, const float* bias, float* C,
                    int M, int N, int K, float alpha) {
        dim3 grid(N / 256, M / 16);
        k_gemm_bf16<<<grid, 128, 0, stream>>>(A, BT, bias, C, M, N, K, alpha);
    };

    // ---- weight conversion (transpose where GEMM needs B^T) ----
    cvt(emb, emb_b, (size_t)VOCAB * DMODEL);
    cvt(mem, mem_b, (size_t)NSLOT * DMODEL);
    cvtT(mem, memT, NSLOT, DMODEL);
    cvtT(Wsp, wspT, DMODEL, DMODEL);
    cvtT(Wrp, wrpT, DMODEL, DMODEL);
    for (int i = 0; i < NLAYER; ++i) {
        cvtT(Wq + (size_t)i * DD, wqT + (size_t)i * DD, DMODEL, DMODEL);
        cvtT(Wk + (size_t)i * DD, wkT + (size_t)i * DD, DMODEL, DMODEL);
        cvtT(Wv + (size_t)i * DD, wvT + (size_t)i * DD, DMODEL, DMODEL);
        cvtT(Wo + (size_t)i * DD, woT + (size_t)i * DD, DMODEL, DMODEL);
        cvtT(Wg + (size_t)i * DMODEL * DFF, wgT + (size_t)i * DMODEL * DFF, DMODEL, DFF);
        cvtT(Wu + (size_t)i * DMODEL * DFF, wuT + (size_t)i * DMODEL * DFF, DMODEL, DFF);
        cvtT(Wd + (size_t)i * DFF * DMODEL, wdT + (size_t)i * DFF * DMODEL, DFF, DMODEL);
    }

    // ---- embed + rope ----
    k_embed_rope<<<NTOK, 256, 0, stream>>>(ids, emb, xf, xb);

    // ---- transformer layers ----
    for (int i = 0; i < NLAYER; ++i) {
        gemm(xb, wqT + (size_t)i * DD, nullptr, qf, NTOK, DMODEL, DMODEL, 1.f);
        gemm(xb, wkT + (size_t)i * DD, nullptr, kf, NTOK, DMODEL, DMODEL, 1.f);
        gemm(xb, wvT + (size_t)i * DD, nullptr, vf, NTOK, DMODEL, DMODEL, 1.f);
        k_attn<<<(NTOK * NHEAD) / 256, 256, 0, stream>>>(qf, kf, vf, obf);
        gemm(obf, woT + (size_t)i * DD, bo + (size_t)i * DMODEL, tf, NTOK, DMODEL, DMODEL, 1.f);
        k_resid_rmsnorm<<<NTOK, 256, 0, stream>>>(xf, tf, n1 + (size_t)i * DMODEL, xf, xb);
        gemm(xb, wgT + (size_t)i * DMODEL * DFF, nullptr, gf, NTOK, DFF, DMODEL, 1.f);
        gemm(xb, wuT + (size_t)i * DMODEL * DFF, nullptr, uf, NTOK, DFF, DMODEL, 1.f);
        k_silu_mul<<<4096, 256, 0, stream>>>(gf, uf, ebf, (size_t)NTOK * DFF);
        gemm(ebf, wdT + (size_t)i * DFF * DMODEL, nullptr, tf, NTOK, DMODEL, DFF, 1.f);
        k_resid_rmsnorm<<<NTOK, 256, 0, stream>>>(xf, tf, n2 + (size_t)i * DMODEL, xf, xb);
    }

    // ---- slot memory read ----
    gemm(xb, wspT, bsp, tf, NTOK, DMODEL, DMODEL, 1.f);          // q = x@Wsp + bsp
    cvt(tf, obf, (size_t)NTOK * DMODEL);                         // q -> bf16
    gemm(obf, mem_b, nullptr, scf, NTOK, NSLOT, DMODEL, 0.03125f); // q@mem^T * D^-0.5
    k_softmax_bf16<<<NTOK, 256, 0, stream>>>(scf, scb, NSLOT);
    gemm(scb, memT, nullptr, qf, NTOK, DMODEL, NSLOT, 1.f);      // r = attn @ mem
    cvt(qf, obf, (size_t)NTOK * DMODEL);                         // r -> bf16
    gemm(obf, wrpT, brp, tf, NTOK, DMODEL, DMODEL, 1.f);         // r@Wrp + brp
    k_resid_rmsnorm<<<NTOK, 256, 0, stream>>>(xf, tf, nout, kf, xb); // h (bf16 in xb)

    // ---- logits = h @ emb^T ----
    gemm(xb, emb_b, nullptr, (float*)d_out, NTOK, VOCAB, DMODEL, 1.f);
}
